// BrainNetworkFusionModel_46703474376901
// MI455X (gfx1250) — matrix operand order repaired and checked by reading, not compiled, not run on hardware
//
#include <hip/hip_runtime.h>
#include <hip/hip_bf16.h>
#include <math.h>

// ---------------------------------------------------------------------------
// CDNA5 (gfx1250) BrainNetworkFusion forward pass.
// Dense matmuls: operands pre-packed into v_wmma_f32_16x16x32_f16 fragment
// layout (f16, 32 B per lane) so the GEMM hot loop is pure
// global_load_b128 + v_wmma (4 row-tiles per wave, B fragment reused 4x).
// Edge aggregation uses f32 global atomics (activations are 26 MB << 192 MB L2).
// ---------------------------------------------------------------------------

typedef __attribute__((ext_vector_type(16))) _Float16 v16h;
typedef __attribute__((ext_vector_type(8)))  float    v8f;

#define BN_SCALE_F 0.9999950000374997f  // 1/sqrt(1 + 1e-5)

// ------------------------------ utility kernels ----------------------------

__global__ void fill_kernel(float* __restrict__ p, float v, int n) {
  int i = blockIdx.x * blockDim.x + threadIdx.x;
  if (i < n) p[i] = v;
}

__global__ void deg_edges_kernel(const int* __restrict__ dst,
                                 const float* __restrict__ w,
                                 float* __restrict__ deg, int E) {
  int e = blockIdx.x * blockDim.x + threadIdx.x;
  if (e < E) atomicAdd(&deg[dst[e]], w[e]);
}

__global__ void deg_to_dinv_kernel(float* __restrict__ d, int n) {
  int i = blockIdx.x * blockDim.x + threadIdx.x;
  if (i < n) {
    float v = d[i];
    d[i] = (v > 0.0f) ? rsqrtf(v) : 0.0f;
  }
}

__global__ void edge_norm_kernel(const int* __restrict__ src,
                                 const int* __restrict__ dst,
                                 const float* __restrict__ w,
                                 const float* __restrict__ dinv,
                                 float* __restrict__ wn, int E) {
  int e = blockIdx.x * blockDim.x + threadIdx.x;
  if (e < E) wn[e] = dinv[src[e]] * w[e] * dinv[dst[e]];
}

// agg[i][:] = dinv[i]^2 * m[i][:]   (self-loop term; initializes agg)
__global__ void self_init_kernel(const float* __restrict__ m,
                                 const float* __restrict__ dinv,
                                 float* __restrict__ agg, int N) {
  int idx = blockIdx.x * blockDim.x + threadIdx.x;  // N*32 threads (4 cols each)
  if (idx < N * 32) {
    int i  = idx >> 5;
    int c4 = (idx & 31) << 2;
    float s = dinv[i];
    s = s * s;
    float4 mv = *(const float4*)(m + (size_t)i * 128 + c4);
    float4 o  = { s * mv.x, s * mv.y, s * mv.z, s * mv.w };
    *(float4*)(agg + (size_t)i * 128 + c4) = o;
  }
}

// agg[dst[e]][:] += wn[e] * m[src[e]][:]
__global__ void scatter_kernel(const int* __restrict__ src,
                               const int* __restrict__ dst,
                               const float* __restrict__ wn,
                               const float* __restrict__ m,
                               float* __restrict__ agg, int E) {
  long long idx = (long long)blockIdx.x * blockDim.x + threadIdx.x;  // E*32
  if (idx < (long long)E * 32) {
    int e  = (int)(idx >> 5);
    int c4 = ((int)idx & 31) << 2;
    float w = wn[e];
    int s = src[e], d = dst[e];
    float4 mv = *(const float4*)(m + (size_t)s * 128 + c4);
    float* a = agg + (size_t)d * 128 + c4;
    atomicAdd(a + 0, w * mv.x);
    atomicAdd(a + 1, w * mv.y);
    atomicAdd(a + 2, w * mv.z);
    atomicAdd(a + 3, w * mv.w);
  }
}

// out = relu((agg + bias) * (g*BN_SCALE) + bb)   (Hd power of two)
__global__ void bias_bn_relu_kernel(const float* __restrict__ agg,
                                    const float* __restrict__ b,
                                    const float* __restrict__ g,
                                    const float* __restrict__ bb,
                                    float* __restrict__ out, int total, int Hd) {
  int i = blockIdx.x * blockDim.x + threadIdx.x;
  if (i < total) {
    int c = i & (Hd - 1);
    float v = (agg[i] + b[c]) * (g[c] * BN_SCALE_F) + bb[c];
    out[i] = fmaxf(v, 0.0f);
  }
}

// gate = sigmoid(t + gb); hf = gate*hsc + (1-gate)*hfc
__global__ void gate_fuse_kernel(const float* __restrict__ t,
                                 const float* __restrict__ gb,
                                 const float* __restrict__ hsc,
                                 const float* __restrict__ hfc,
                                 float* __restrict__ hf, int total, int Hd) {
  int i = blockIdx.x * blockDim.x + threadIdx.x;
  if (i < total) {
    int c = i & (Hd - 1);
    float z = t[i] + gb[c];
    float gate = 1.0f / (1.0f + expf(-z));
    hf[i] = gate * hsc[i] + (1.0f - gate) * hfc[i];
  }
}

__global__ void pool_acc_kernel(const float* __restrict__ hf,
                                const int* __restrict__ batch,
                                float* __restrict__ sums,
                                float* __restrict__ cnts, int N) {
  int idx = blockIdx.x * blockDim.x + threadIdx.x;  // N*32
  if (idx < N * 32) {
    int i  = idx >> 5;
    int c4 = (idx & 31) << 2;
    int bi = batch[i];
    float4 v = *(const float4*)(hf + (size_t)i * 128 + c4);
    float* s = sums + (size_t)bi * 128 + c4;
    atomicAdd(s + 0, v.x);
    atomicAdd(s + 1, v.y);
    atomicAdd(s + 2, v.z);
    atomicAdd(s + 3, v.w);
    if (c4 == 0) atomicAdd(&cnts[bi], 1.0f);
  }
}

__global__ void pool_fin_kernel(const float* __restrict__ sums,
                                const float* __restrict__ cnts,
                                float* __restrict__ hg, int total, int Hd) {
  int i = blockIdx.x * blockDim.x + threadIdx.x;
  if (i < total) {
    int g = i / Hd;
    hg[i] = sums[i] / fmaxf(cnts[g], 1.0f);
  }
}

__global__ void final_fc_kernel(const float* __restrict__ z2,
                                const float* __restrict__ W3,
                                const float* __restrict__ b3,
                                float* __restrict__ out, int G, int Din, int C) {
  int idx = blockIdx.x * blockDim.x + threadIdx.x;
  if (idx < G * C) {
    int g = idx / C, c = idx % C;
    float acc = b3[c];
    for (int k = 0; k < Din; ++k) acc += z2[(size_t)g * Din + k] * W3[(size_t)k * C + c];
    out[idx] = acc;
  }
}

// --------------------------- fragment pack kernels -------------------------
// A-fragment layout (ISA 7.12.2, 16-bit A 16x32): lane l -> row m=l&15,
// half g=l>>4; halves t=0..7 -> K = g*8+t, t=8..15 -> K = 16+g*8+(t-8).
// Packed as Ap[rt][s][lane][16 halves] (32 B contiguous per lane).
__global__ void pack_a_kernel(const float* __restrict__ A, _Float16* __restrict__ Ap,
                              long long nfrag /* (M/16)*ksteps */, int K, int ksteps) {
  long long f = (long long)blockIdx.x * blockDim.x + threadIdx.x;  // nfrag*32
  if (f < nfrag * 32) {
    int lane = (int)(f & 31);
    long long rs = f >> 5;
    int s = (int)(rs % ksteps);
    long long rt = rs / ksteps;
    int m = lane & 15, g = lane >> 4;
    const float* row = A + (size_t)(rt * 16 + m) * K;
    int kb = s * 32 + g * 8;
    v16h o;
#pragma unroll
    for (int t = 0; t < 8; ++t) {
      int k0 = kb + t, k1 = kb + 16 + t;
      o[t]     = (_Float16)((k0 < K) ? row[k0] : 0.0f);
      o[t + 8] = (_Float16)((k1 < K) ? row[k1] : 0.0f);
    }
    *((v16h*)Ap + f) = o;
  }
}

// B-fragment layout (32x16): lane l -> col n=l&15, half g=l>>4;
// halves t=0..7 -> K = g*8+t, t=8..15 -> K = 16+g*8+(t-8). Packed as Wp[s][ntile][lane][16 halves].
__global__ void pack_w_kernel(const float* __restrict__ W, _Float16* __restrict__ Wp,
                              int K, int Nout, int ksteps, int ntiles) {
  int f = blockIdx.x * blockDim.x + threadIdx.x;  // ksteps*ntiles*32
  if (f < ksteps * ntiles * 32) {
    int lane = f & 31;
    int sn = f >> 5;
    int nt = sn % ntiles;
    int s  = sn / ntiles;
    int g = lane >> 4, ncol = lane & 15;
    int n = nt * 16 + ncol;
    v16h o;
#pragma unroll
    for (int t = 0; t < 16; ++t) {
      int kk = s * 32 + ((t < 8) ? (g * 8 + t) : (16 + g * 8 + (t - 8)));
      o[t] = (_Float16)((kk < K) ? W[(size_t)kk * Nout + n] : 0.0f);
    }
    *((v16h*)Wp + f) = o;
  }
}

// ------------------------------ WMMA GEMM ----------------------------------
// out[M,Nout] = A @ W from pre-packed fragments. One wave per (4 row-tiles x
// 1 col-tile); block = 32*(Nout/16) threads covers 64 rows x Nout cols.
//   mode 0: out = acc        mode 1: out += acc
//   mode 2: out = acc+bias   mode 3: out = relu((acc+bias)*bn_g*SCALE + bn_b)
__global__ __launch_bounds__(256)
void gemm_wmma_kernel(const _Float16* __restrict__ Ap, const _Float16* __restrict__ Wp,
                      const float* __restrict__ bias, const float* __restrict__ bn_g,
                      const float* __restrict__ bn_b, float* __restrict__ out,
                      int ksteps, int Nout, int ntiles, int mode) {
  const int lane = threadIdx.x & 31;
  const int nt   = threadIdx.x >> 5;     // col tile
  const int rt0  = blockIdx.x * 4;       // first of 4 row tiles

  const v16h* Av = (const v16h*)Ap;
  const v16h* Bv = (const v16h*)Wp;

  v8f acc0 = {}, acc1 = {}, acc2 = {}, acc3 = {};
  for (int s = 0; s < ksteps; ++s) {
    v16h b  = Bv[((size_t)s * ntiles + nt) * 32 + lane];
    v16h a0 = Av[((size_t)(rt0 + 0) * ksteps + s) * 32 + lane];
    v16h a1 = Av[((size_t)(rt0 + 1) * ksteps + s) * 32 + lane];
    v16h a2 = Av[((size_t)(rt0 + 2) * ksteps + s) * 32 + lane];
    v16h a3 = Av[((size_t)(rt0 + 3) * ksteps + s) * 32 + lane];
    acc0 = __builtin_amdgcn_wmma_f32_16x16x32_f16(false, a0, false, b, (short)0, acc0, false, false);
    acc1 = __builtin_amdgcn_wmma_f32_16x16x32_f16(false, a1, false, b, (short)0, acc1, false, false);
    acc2 = __builtin_amdgcn_wmma_f32_16x16x32_f16(false, a2, false, b, (short)0, acc2, false, false);
    acc3 = __builtin_amdgcn_wmma_f32_16x16x32_f16(false, a3, false, b, (short)0, acc3, false, false);
  }

  // C/D layout: VGPR j -> row = gsel*8 + j, col = ncol (within 16x16 tile)
  const int gsel = lane >> 4;
  const int ncol = lane & 15;
  const int c = nt * 16 + ncol;

  auto epi = [&](v8f a, int rtile) {
#pragma unroll
    for (int j = 0; j < 8; ++j) {
      const int r = rtile * 16 + gsel * 8 + j;
      float v = a[j];
      if (mode == 1) {
        v += out[(size_t)r * Nout + c];
      } else if (mode == 2) {
        v += bias[c];
      } else if (mode == 3) {
        v = (v + bias[c]) * (bn_g[c] * BN_SCALE_F) + bn_b[c];
        v = fmaxf(v, 0.0f);
      }
      out[(size_t)r * Nout + c] = v;
    }
  };
  epi(acc0, rt0 + 0);
  epi(acc1, rt0 + 1);
  epi(acc2, rt0 + 2);
  epi(acc3, rt0 + 3);
}

// ------------------------------ host driver --------------------------------

extern "C" void kernel_launch(void* const* d_in, const int* in_sizes, int n_in,
                              void* d_out, int out_size, void* d_ws, size_t ws_size,
                              hipStream_t stream) {
  (void)n_in; (void)ws_size;

  const float* x     = (const float*)d_in[0];
  const int*   ei_sc = (const int*)d_in[1];
  const float* ew_sc = (const float*)d_in[2];
  const int*   ei_fc = (const int*)d_in[3];
  const float* ew_fc = (const float*)d_in[4];
  const int*   batch = (const int*)d_in[5];

  const int N  = in_sizes[5];           // 51200
  const int E  = in_sizes[2];           // 1638400
  const int K0 = in_sizes[0] / N;       // 400
  const int H  = 128;
  const int D  = 64;
  const int C  = 2;
  const int G  = out_size / C;          // 128

  // params (dict order)
  int pi = 6;
  auto P = [&](void) { return (const float*)d_in[pi++]; };
  const float* te_W1 = P(); const float* te_b1 = P();
  const float* te_bn_g = P(); const float* te_bn_b = P();
  const float* te_W2 = P(); const float* te_b2 = P();
  const float* sc_W0 = P(); const float* sc_b0 = P();
  const float* sc_W1 = P(); const float* sc_b1 = P();
  const float* fc_W0 = P(); const float* fc_b0 = P();
  const float* fc_W1 = P(); const float* fc_b1 = P();
  const float* bn_sc_g0 = P(); const float* bn_sc_b0 = P();
  const float* bn_sc_g1 = P(); const float* bn_sc_b1 = P();
  const float* bn_fc_g0 = P(); const float* bn_fc_b0 = P();
  const float* bn_fc_g1 = P(); const float* bn_fc_b1 = P();
  const float* gate_W = P(); const float* gate_b = P();
  const float* pr_W1 = P(); const float* pr_b1 = P();
  const float* pr_bn1_g = P(); const float* pr_bn1_b = P();
  const float* pr_W2 = P(); const float* pr_b2 = P();
  const float* pr_bn2_g = P(); const float* pr_bn2_b = P();
  const float* pr_W3 = P(); const float* pr_b3 = P();

  const int* src_s = ei_sc;      const int* dst_s = ei_sc + E;
  const int* src_f = ei_fc;      const int* dst_f = ei_fc + E;

  // workspace carve-up (256B aligned)
  size_t off = 0;
  auto alloc = [&](size_t bytes) -> char* {
    off = (off + 255) & ~(size_t)255;
    char* p = (char*)d_ws + off;
    off += bytes;
    return p;
  };
  float* h1     = (float*)alloc((size_t)N * H * 4);   // te hidden; reused as m / gate tmp
  float* henc   = (float*)alloc((size_t)N * D * 4);
  float* hsc    = (float*)alloc((size_t)N * H * 4);
  float* hfc    = (float*)alloc((size_t)N * H * 4);
  float* agg    = (float*)alloc((size_t)N * H * 4);   // also h_fused
  float* wn_s   = (float*)alloc((size_t)E * 4);
  float* wn_f   = (float*)alloc((size_t)E * 4);
  float* dinv_s = (float*)alloc((size_t)N * 4);
  float* dinv_f = (float*)alloc((size_t)N * 4);
  float* sums   = (float*)alloc((size_t)G * H * 4);
  float* cnts   = (float*)alloc((size_t)G * 4);
  float* hg     = (float*)alloc((size_t)G * H * 4);
  float* z1     = (float*)alloc((size_t)G * H * 4);
  float* z2     = (float*)alloc((size_t)G * D * 4);
  // packed fragment scratch (reused per GEMM; max = x with K=400 -> 13 ksteps)
  const int ksteps_max = (K0 + 31) / 32;
  _Float16* Ap = (_Float16*)alloc((size_t)(N / 16) * ksteps_max * 512 * 2);
  _Float16* Wp = (_Float16*)alloc((size_t)ksteps_max * (H / 16) * 512 * 2);
  float* m = h1;  // alias: h1 free after henc is computed

  auto nb = [](long long n) { return (unsigned)((n + 255) / 256); };

  // ---- gcn_norm for both edge sets (deg init 1.0 covers self-loop) ----
  fill_kernel<<<nb(N), 256, 0, stream>>>(dinv_s, 1.0f, N);
  fill_kernel<<<nb(N), 256, 0, stream>>>(dinv_f, 1.0f, N);
  deg_edges_kernel<<<nb(E), 256, 0, stream>>>(dst_s, ew_sc, dinv_s, E);
  deg_edges_kernel<<<nb(E), 256, 0, stream>>>(dst_f, ew_fc, dinv_f, E);
  deg_to_dinv_kernel<<<nb(N), 256, 0, stream>>>(dinv_s, N);
  deg_to_dinv_kernel<<<nb(N), 256, 0, stream>>>(dinv_f, N);
  edge_norm_kernel<<<nb(E), 256, 0, stream>>>(src_s, dst_s, ew_sc, dinv_s, wn_s, E);
  edge_norm_kernel<<<nb(E), 256, 0, stream>>>(src_f, dst_f, ew_fc, dinv_f, wn_f, E);

  // pack + GEMM helper
  auto run_gemm = [&](const float* A, const float* W, const float* bias,
                      const float* gg, const float* bb, float* out,
                      int M, int K, int Nout, int mode) {
    const int ksteps = (K + 31) / 32;
    const int ntiles = Nout / 16;
    const long long nfragA = (long long)(M / 16) * ksteps;
    pack_a_kernel<<<nb(nfragA * 32), 256, 0, stream>>>(A, Ap, nfragA, K, ksteps);
    pack_w_kernel<<<nb((long long)ksteps * ntiles * 32), 256, 0, stream>>>(
        W, Wp, K, Nout, ksteps, ntiles);
    dim3 grid(M / 64);
    dim3 block(32 * ntiles);
    gemm_wmma_kernel<<<grid, block, 0, stream>>>(Ap, Wp, bias, gg, bb, out,
                                                 ksteps, Nout, ntiles, mode);
  };

  // ---- TemporalEncoder ----
  run_gemm(x,  te_W1, te_b1, te_bn_g, te_bn_b, h1,   N, K0, H, 3);
  run_gemm(h1, te_W2, te_b2, nullptr, nullptr, henc, N, H,  D, 2);
  // h1 now free -> reused as m

  // ---- GCN layer helper ----
  auto gcn_layer = [&](const float* hin, int Kin, const float* W, const float* bias,
                       const float* gg, const float* bb,
                       const int* src, const int* dst, const float* wn,
                       const float* dinv, float* hout) {
    run_gemm(hin, W, nullptr, nullptr, nullptr, m, N, Kin, H, 0);  // m = hin @ W
    self_init_kernel<<<nb((long long)N * 32), 256, 0, stream>>>(m, dinv, agg, N);
    scatter_kernel<<<nb((long long)E * 32), 256, 0, stream>>>(src, dst, wn, m, agg, E);
    bias_bn_relu_kernel<<<nb((long long)N * H), 256, 0, stream>>>(
        agg, bias, gg, bb, hout, N * H, H);
  };

  // SC branch
  gcn_layer(henc, D, sc_W0, sc_b0, bn_sc_g0, bn_sc_b0, src_s, dst_s, wn_s, dinv_s, hsc);
  gcn_layer(hsc,  H, sc_W1, sc_b1, bn_sc_g1, bn_sc_b1, src_s, dst_s, wn_s, dinv_s, hsc);
  // FC branch
  gcn_layer(henc, D, fc_W0, fc_b0, bn_fc_g0, bn_fc_b0, src_f, dst_f, wn_f, dinv_f, hfc);
  gcn_layer(hfc,  H, fc_W1, fc_b1, bn_fc_g1, bn_fc_b1, src_f, dst_f, wn_f, dinv_f, hfc);

  // ---- gated fusion: gate = sigmoid([hsc|hfc] @ gate_W + gate_b) ----
  run_gemm(hsc, gate_W,                 nullptr, nullptr, nullptr, m, N, H, H, 0);
  run_gemm(hfc, gate_W + (size_t)H * H, nullptr, nullptr, nullptr, m, N, H, H, 1);
  gate_fuse_kernel<<<nb((long long)N * H), 256, 0, stream>>>(
      m, gate_b, hsc, hfc, agg /* h_fused */, N * H, H);

  // ---- global mean pool ----
  fill_kernel<<<nb(G * H), 256, 0, stream>>>(sums, 0.0f, G * H);
  fill_kernel<<<nb(G), 256, 0, stream>>>(cnts, 0.0f, G);
  pool_acc_kernel<<<nb((long long)N * 32), 256, 0, stream>>>(agg, batch, sums, cnts, N);
  pool_fin_kernel<<<nb(G * H), 256, 0, stream>>>(sums, cnts, hg, G * H, H);

  // ---- predictor ----
  run_gemm(hg, pr_W1, pr_b1, pr_bn1_g, pr_bn1_b, z1, G, H, H, 3);
  run_gemm(z1, pr_W2, pr_b2, pr_bn2_g, pr_bn2_b, z2, G, H, D, 3);
  final_fc_kernel<<<nb(G * C), 256, 0, stream>>>(z2, pr_W3, pr_b3, (float*)d_out, G, D, C);
}
